// SelfAttentiveLBLBiLM_14456859918901
// MI455X (gfx1250) — compile-verified, matrix operand order checked by reading, not run on hardware
//
#include <hip/hip_runtime.h>
#include <hip/hip_bf16.h>

typedef __bf16 bf16_t;
typedef __attribute__((ext_vector_type(16))) __bf16 v16bf;
typedef __attribute__((ext_vector_type(8)))  float  v8f;

#define B_   16
#define S_   512
#define D_   512
#define H_   8
#define L_   2
#define W_   16
#define NHW_ 2
#define P_   (S_ + 2 * W_)
#define DK_  (D_ / H_)
#define KK_  D_   // reduction dim for all GEMMs (=512)

// ---------------- f32 -> bf16 conversion (weights) ----------------
__global__ void __launch_bounds__(256) k_cvt(const float* __restrict__ src,
                                             bf16_t* __restrict__ dst, int n) {
  int i = blockIdx.x * 256 + threadIdx.x;
  if (i < n) dst[i] = (bf16_t)src[i];
}

// ---------------- build padded bf16 activations (B,P,D) ----------------
__global__ void __launch_bounds__(256) k_pad(const float* __restrict__ cur,
                                             const float* __restrict__ fpad,
                                             const float* __restrict__ bpad,
                                             bf16_t* __restrict__ xp) {
  size_t i = (size_t)blockIdx.x * 256 + threadIdx.x;   // over B*P*D
  int d = (int)(i % D_);
  size_t r = i / D_;
  int p = (int)(r % P_);
  int b = (int)(r / P_);
  float v;
  if (p < W_)              v = fpad[p * D_ + d];
  else if (p >= W_ + S_)   v = bpad[(p - W_ - S_) * D_ + d];
  else                     v = cur[((size_t)b * S_ + (p - W_)) * D_ + d];
  xp[i] = (bf16_t)v;
}

// ---------------- bf16 WMMA GEMM: C(M,N) = A(M,K) * W(N,K)^T + bias ----------------
// block = 256 threads = 8 waves as 4(M) x 2(N); wave tile 32(M) x 64(N);
// K step 32, register double-buffered fragments, K = 512 compile-time.
#define WMMA_BF16(ACC, AV, BV) \
  ACC = __builtin_amdgcn_wmma_f32_16x16x32_bf16(false, AV, false, BV, (short)0, ACC, false, false)

template <typename OutT>
__global__ void __launch_bounds__(256) k_gemm(const bf16_t* __restrict__ A,
                                              const bf16_t* __restrict__ Wt,
                                              const float* __restrict__ bias,
                                              OutT* __restrict__ C,
                                              int M, int N) {
  const int lane = threadIdx.x & 31;
  const int wave = threadIdx.x >> 5;
  const int mw = wave & 3;        // 4 M-waves
  const int nw = wave >> 2;       // 2 N-waves
  const int lo = lane & 15;
  const int hi = lane >> 4;
  const int m0 = blockIdx.x * 128 + mw * 32;
  const int n0 = blockIdx.y * 128 + nw * 64;
  (void)M;

  // A fragments: lane -> row, K-slice [hi*16, hi*16+16)
  const bf16_t* ap0 = A + (size_t)(m0 + lo) * KK_ + hi * 16;
  const bf16_t* ap1 = A + (size_t)(m0 + 16 + lo) * KK_ + hi * 16;
  // B fragments: lane -> weight row n (= output column), same K-slice
  const bf16_t* bp = Wt + (size_t)(n0 + lo) * KK_ + hi * 16;

  v8f acc[2][4];
#pragma unroll
  for (int ms = 0; ms < 2; ++ms)
#pragma unroll
    for (int t = 0; t < 4; ++t) acc[ms][t] = (v8f){};

  v16bf a0 = *(const v16bf*)(ap0);
  v16bf a1 = *(const v16bf*)(ap1);
  v16bf b0 = *(const v16bf*)(bp);
  v16bf b1 = *(const v16bf*)(bp + (size_t)16 * KK_);
  v16bf b2 = *(const v16bf*)(bp + (size_t)32 * KK_);
  v16bf b3 = *(const v16bf*)(bp + (size_t)48 * KK_);

#define WMMA8()             \
  WMMA_BF16(acc[0][0], a0, b0); \
  WMMA_BF16(acc[0][1], a0, b1); \
  WMMA_BF16(acc[0][2], a0, b2); \
  WMMA_BF16(acc[0][3], a0, b3); \
  WMMA_BF16(acc[1][0], a1, b0); \
  WMMA_BF16(acc[1][1], a1, b1); \
  WMMA_BF16(acc[1][2], a1, b2); \
  WMMA_BF16(acc[1][3], a1, b3)

  for (int k0 = 32; k0 < KK_; k0 += 32) {
    // prefetch future K-steps into WGP$ (global_prefetch_b8)
    __builtin_prefetch(ap0 + k0 + 96, 0, 3);
    __builtin_prefetch(ap1 + k0 + 96, 0, 3);
    __builtin_prefetch(bp + k0 + 96, 0, 3);
    // next-iteration fragments issued BEFORE this iteration's WMMAs
    v16bf na0 = *(const v16bf*)(ap0 + k0);
    v16bf na1 = *(const v16bf*)(ap1 + k0);
    v16bf nb0 = *(const v16bf*)(bp + k0);
    v16bf nb1 = *(const v16bf*)(bp + (size_t)16 * KK_ + k0);
    v16bf nb2 = *(const v16bf*)(bp + (size_t)32 * KK_ + k0);
    v16bf nb3 = *(const v16bf*)(bp + (size_t)48 * KK_ + k0);
    WMMA8();
    a0 = na0; a1 = na1; b0 = nb0; b1 = nb1; b2 = nb2; b3 = nb3;
  }
  WMMA8();   // tail K-step

  // C/D layout: element (m = m0 + ms*16 + hi*8 + r, n = n0 + t*16 + lo)
#pragma unroll
  for (int t = 0; t < 4; ++t) {
    const int n = n0 + t * 16 + lo;
    const float bv = bias[n];
#pragma unroll
    for (int ms = 0; ms < 2; ++ms) {
#pragma unroll
      for (int r = 0; r < 8; ++r) {
        const size_t m = (size_t)(m0 + ms * 16 + hi * 8 + r);
        C[m * (size_t)N + n] = (OutT)(acc[ms][t][r] + bv);
      }
    }
  }
}

// ---------------- banded attention: one wave per (b,h,query) ----------------
// q,k,v bf16 (B,P,D) head-interleaved; o bf16 (B,P,D). 18 keys per query.
__global__ void __launch_bounds__(256) k_attn(const bf16_t* __restrict__ q,
                                              const bf16_t* __restrict__ kk,
                                              const bf16_t* __restrict__ vv,
                                              bf16_t* __restrict__ o, int fwd) {
  const int lane = threadIdx.x & 31;
  const int wave = threadIdx.x >> 5;
  size_t gq = (size_t)blockIdx.x * 8 + wave;   // global query id over B*H*P
  const int p = (int)(gq % P_);
  size_t bh = gq / P_;
  const int h = (int)(bh % H_);
  const int b = (int)(bh / H_);

  const bf16_t* qrow = q + ((size_t)b * P_ + p) * D_ + h * DK_;

  const int j = fwd ? (p - (W_ + 1) + lane) : (p + lane);
  const bool valid = (lane <= W_ + 1) && (j >= 0) && (j < P_);

  float s = -1e30f;
  if (valid) {
    const bf16_t* krow = kk + ((size_t)b * P_ + j) * D_ + h * DK_;
    float acc = 0.0f;
#pragma unroll 8
    for (int d = 0; d < DK_; ++d) acc += (float)qrow[d] * (float)krow[d];
    s = acc * 0.125f;   // 1/sqrt(64)
  }

  // wave32 max-reduce
  float mx = s;
#pragma unroll
  for (int off = 16; off > 0; off >>= 1) mx = fmaxf(mx, __shfl_xor(mx, off, 32));
  float e = valid ? __expf(s - mx) : 0.0f;
  float sum = e;
#pragma unroll
  for (int off = 16; off > 0; off >>= 1) sum += __shfl_xor(sum, off, 32);
  const float pr = e / sum;

  // P*V: each lane owns dk columns {lane, lane+32}
  float a0 = 0.0f, a1 = 0.0f;
#pragma unroll
  for (int l = 0; l <= W_ + 1; ++l) {
    const float pj = __shfl(pr, l, 32);
    const int jj = fwd ? (p - (W_ + 1) + l) : (p + l);
    if (jj >= 0 && jj < P_) {
      const bf16_t* vrow = vv + ((size_t)b * P_ + jj) * D_ + h * DK_;
      a0 += pj * (float)vrow[lane];
      a1 += pj * (float)vrow[lane + 32];
    }
  }
  bf16_t* orow = o + ((size_t)b * P_ + p) * D_ + h * DK_;
  orow[lane] = (bf16_t)a0;
  orow[lane + 32] = (bf16_t)a1;
}

// ---------------- relative-position combine: fa (B,P,D) -> x (B,S,D) ----------------
__global__ void __launch_bounds__(256) k_relw(const float* __restrict__ fa,
                                              const float* __restrict__ rw,
                                              float* __restrict__ x,
                                              bf16_t* __restrict__ xb, int fwd) {
  size_t i = (size_t)blockIdx.x * 256 + threadIdx.x;   // over B*S*D
  int d = (int)(i % D_);
  size_t r = i / D_;
  int s = (int)(r % S_);
  int b = (int)(r / S_);
  const float* base = fa + (size_t)b * P_ * D_;
  float acc = base[(size_t)(W_ + s) * D_ + d];
#pragma unroll
  for (int k = 0; k <= W_; ++k) {
    const int pos = fwd ? (k + s) : (W_ + k + s);
    acc += rw[k] * base[(size_t)pos * D_ + d];
  }
  x[i] = acc;
  xb[i] = (bf16_t)acc;
}

// ---------------- highway gate: x = sig(g)*x + (1-sig(g))*relu(nl) ----------------
__global__ void __launch_bounds__(256) k_highway(const float* __restrict__ proj,
                                                 float* __restrict__ x,
                                                 bf16_t* __restrict__ xb) {
  size_t i = (size_t)blockIdx.x * 256 + threadIdx.x;   // over B*S*D
  int d = (int)(i % D_);
  size_t row = i / D_;
  const float nl = proj[row * (2 * D_) + d];
  const float g  = proj[row * (2 * D_) + D_ + d];
  const float sig = 1.0f / (1.0f + __expf(-g));
  const float nx = sig * x[i] + (1.0f - sig) * fmaxf(nl, 0.0f);
  x[i] = nx;
  xb[i] = (bf16_t)nx;
}

// ---------------- concat [fcur, bcur] into out[layer] ----------------
__global__ void __launch_bounds__(256) k_concat(const float* __restrict__ f,
                                                const float* __restrict__ bq,
                                                float* __restrict__ out) {
  size_t i = (size_t)blockIdx.x * 256 + threadIdx.x;   // over B*S*D
  int d = (int)(i % D_);
  size_t row = i / D_;
  out[row * (2 * D_) + d] = f[i];
  out[row * (2 * D_) + D_ + d] = bq[i];
}

extern "C" void kernel_launch(void* const* d_in, const int* in_sizes, int n_in,
                              void* d_out, int out_size, void* d_ws, size_t ws_size,
                              hipStream_t stream) {
  (void)in_sizes; (void)n_in; (void)out_size; (void)ws_size;

  const float* inputs  = (const float*)d_in[0];
  const float* f_lin_w = (const float*)d_in[1];
  const float* f_lin_b = (const float*)d_in[2];
  const float* b_lin_w = (const float*)d_in[3];
  const float* b_lin_b = (const float*)d_in[4];
  const float* f_pad   = (const float*)d_in[5];
  const float* b_pad   = (const float*)d_in[6];
  const float* f_hw_w  = (const float*)d_in[7];
  const float* f_hw_b  = (const float*)d_in[8];
  const float* b_hw_w  = (const float*)d_in[9];
  const float* b_hw_b  = (const float*)d_in[10];
  const float* f_relw  = (const float*)d_in[11];
  const float* b_relw  = (const float*)d_in[12];
  float* out = (float*)d_out;

  // -------- workspace layout (~155 MB) --------
  char* ws = (char*)d_ws;
  size_t off = 0;
  auto alloc = [&](size_t bytes) -> void* {
    void* p = ws + off;
    off += (bytes + 255) & ~(size_t)255;
    return p;
  };
  float*  fcur  = (float*)alloc((size_t)B_ * S_ * D_ * 4);
  float*  bcur  = (float*)alloc((size_t)B_ * S_ * D_ * 4);
  bf16_t* flinw = (bf16_t*)alloc((size_t)L_ * 4 * D_ * D_ * 2);
  bf16_t* blinw = (bf16_t*)alloc((size_t)L_ * 4 * D_ * D_ * 2);
  bf16_t* fhww  = (bf16_t*)alloc((size_t)L_ * NHW_ * 2 * D_ * D_ * 2);
  bf16_t* bhww  = (bf16_t*)alloc((size_t)L_ * NHW_ * 2 * D_ * D_ * 2);
  bf16_t* xpad  = (bf16_t*)alloc((size_t)B_ * P_ * D_ * 2);
  bf16_t* qb    = (bf16_t*)alloc((size_t)B_ * P_ * D_ * 2);
  bf16_t* kb    = (bf16_t*)alloc((size_t)B_ * P_ * D_ * 2);
  bf16_t* vb    = (bf16_t*)alloc((size_t)B_ * P_ * D_ * 2);
  bf16_t* ob    = (bf16_t*)alloc((size_t)B_ * P_ * D_ * 2);
  float*  fa    = (float*)alloc((size_t)B_ * P_ * D_ * 4);
  bf16_t* xb    = (bf16_t*)alloc((size_t)B_ * S_ * D_ * 2);
  float*  proj  = (float*)alloc((size_t)B_ * S_ * 2 * D_ * 4);

  // initial state
  hipMemcpyAsync(fcur, inputs, (size_t)B_ * S_ * D_ * 4, hipMemcpyDeviceToDevice, stream);
  hipMemcpyAsync(bcur, inputs, (size_t)B_ * S_ * D_ * 4, hipMemcpyDeviceToDevice, stream);

  // one-time weight conversion to bf16
  const int nlin = L_ * 4 * D_ * D_;
  const int nhw  = L_ * NHW_ * 2 * D_ * D_;
  k_cvt<<<nlin / 256, 256, 0, stream>>>(f_lin_w, flinw, nlin);
  k_cvt<<<nlin / 256, 256, 0, stream>>>(b_lin_w, blinw, nlin);
  k_cvt<<<nhw / 256, 256, 0, stream>>>(f_hw_w, fhww, nhw);
  k_cvt<<<nhw / 256, 256, 0, stream>>>(b_hw_w, bhww, nhw);

  const int BP = B_ * P_;       // 8704
  const int BS = B_ * S_;       // 8192
  const dim3 gProj(BP / 128, D_ / 128);          // 68 x 4
  const dim3 gHW(BS / 128, (2 * D_) / 128);      // 64 x 8
  const int gElemS = BS * D_ / 256;              // 16384
  const int gElemP = (B_ * P_ * D_) / 256;       // 17408
  const int gAttn  = (B_ * H_ * P_) / 8;         // 8704

  for (int i = 0; i < L_; ++i) {
    for (int dir = 0; dir < 2; ++dir) {
      float* cur = dir ? bcur : fcur;
      const bf16_t* lw  = (dir ? blinw : flinw) + (size_t)i * 4 * D_ * D_;
      const float*  lb  = (dir ? b_lin_b : f_lin_b) + (size_t)i * 4 * D_;
      const bf16_t* hww = (dir ? bhww : fhww) + (size_t)i * NHW_ * 2 * D_ * D_;
      const float*  hwb = (dir ? b_hw_b : f_hw_b) + (size_t)i * NHW_ * 2 * D_;
      const float*  rw  = (dir ? b_relw : f_relw) + (size_t)i * (W_ + 1);
      const int fwd = (dir == 0);

      // both directions pad with f_pad front / b_pad back (per reference)
      k_pad<<<gElemP, 256, 0, stream>>>(cur, f_pad + (size_t)i * W_ * D_,
                                        b_pad + (size_t)i * W_ * D_, xpad);

      k_gemm<bf16_t><<<gProj, 256, 0, stream>>>(xpad, lw + (size_t)0 * D_ * D_, lb + 0 * D_, qb, BP, D_);
      k_gemm<bf16_t><<<gProj, 256, 0, stream>>>(xpad, lw + (size_t)1 * D_ * D_, lb + 1 * D_, kb, BP, D_);
      k_gemm<bf16_t><<<gProj, 256, 0, stream>>>(xpad, lw + (size_t)2 * D_ * D_, lb + 2 * D_, vb, BP, D_);

      k_attn<<<gAttn, 256, 0, stream>>>(qb, kb, vb, ob, fwd);

      k_gemm<float><<<gProj, 256, 0, stream>>>(ob, lw + (size_t)3 * D_ * D_, lb + 3 * D_, fa, BP, D_);

      k_relw<<<gElemS, 256, 0, stream>>>(fa, rw, cur, xb, fwd);

      for (int j = 0; j < NHW_; ++j) {
        k_gemm<float><<<gHW, 256, 0, stream>>>(xb, hww + (size_t)j * 2 * D_ * D_,
                                               hwb + (size_t)j * 2 * D_, proj, BS, 2 * D_);
        k_highway<<<gElemS, 256, 0, stream>>>(proj, cur, xb);
      }
    }
    k_concat<<<gElemS, 256, 0, stream>>>(fcur, bcur, out + (size_t)i * B_ * S_ * 2 * D_);
  }
}
